// InformerModel_86285892977194
// MI455X (gfx1250) — compile-verified, matrix operand order checked by reading, not run on hardware
//
#include <hip/hip_runtime.h>
#include <hip/hip_bf16.h>
#include <math.h>

typedef _Float16 f16;
typedef __attribute__((ext_vector_type(16))) _Float16 v16h;
typedef __attribute__((ext_vector_type(8)))  float    v8f;

static constexpr int Bc = 128, Tc = 1024, INc = 32, Dc = 128, Hc = 8, Lc = 2;
static constexpr int Dh = Dc / Hc;   // 16

// ---------------------------------------------------------------------------
// WMMA fragment load: 16-bit A/B operand, 16x32 (MxK) per the CDNA5 layout:
//   lane&15 = row (M for A, N column for Bt), lane>=16 selects K+8 half,
//   v16h element e -> K = e + (e>=8 ? 8 : 0) + (lane>=16 ? 8 : 0)
// => two contiguous 16-byte loads per fragment.
// ---------------------------------------------------------------------------
union FragU { v16h v; uint4 q[2]; };

__device__ __forceinline__ v16h load_frag_row(const f16* __restrict__ rowp, int k0, int hi) {
  FragU f;
  const int kb = k0 + hi * 8;                          // halves; 16B aligned
  f.q[0] = *reinterpret_cast<const uint4*>(rowp + kb);        // K = kb..kb+7
  f.q[1] = *reinterpret_cast<const uint4*>(rowp + kb + 16);   // K = kb+16..kb+23
  return f.v;
}

template<bool GELU>
__device__ __forceinline__ void store_tile(const v8f& acc, int mbase, int nbase, int N,
                                           const float* __restrict__ bias,
                                           float* __restrict__ Cf, f16* __restrict__ Ch,
                                           int lane) {
  const int hi = lane >> 4, lw = lane & 15;
  const int n  = nbase + lw;                           // C layout: N = lane&15
  const float bv = bias ? bias[n] : 0.0f;
#pragma unroll
  for (int r = 0; r < 8; ++r) {
    const int m = mbase + r + hi * 8;                  // C layout: M = r + 8*(lane>=16)
    float v = acc[r] + bv;
    if (GELU) v = 0.5f * v * (1.0f + erff(v * 0.70710678118654752f));
    const size_t off = (size_t)m * N + n;
    if (Cf) Cf[off] = v;
    if (Ch) Ch[off] = (f16)v;
  }
}

// C(MxN) = act(A(MxKT) @ B(KTxN) + bias).  A row-major f16, Bt = B transposed
// (NxKT row-major) f16.  One wave -> 32x64 super-tile (2x4 WMMA tiles):
// 12 b128 loads per 8 WMMAs; KT compile-time so the K loop fully unrolls
// (immediate address offsets, deep load/WMMA software pipelining).
// GATHER: logical row m maps to A row (m/gU)*gT + gidx[m%gU].
template<int KT, bool GELU, bool GATHER>
__global__ void wmma_gemm_kernel(const f16* __restrict__ A, const f16* __restrict__ Bt,
                                 const float* __restrict__ bias,
                                 float* __restrict__ Cf, f16* __restrict__ Ch,
                                 int M, int N,
                                 const int* __restrict__ gidx, int gU, int gT) {
  const int lane = threadIdx.x;
  const int wid  = blockIdx.x * blockDim.y + threadIdx.y;
  const int nsup = N >> 6;                 // 64-wide N super-tiles
  const int ms   = wid / nsup;
  const int ns   = wid % nsup;
  const int hi   = lane >> 4;
  const int lw   = lane & 15;

  const int r0 = ms * 32 + lw;
  const int r1 = r0 + 16;
  int ar0, ar1;
  if (GATHER) {
    ar0 = (r0 / gU) * gT + gidx[r0 % gU];
    ar1 = (r1 / gU) * gT + gidx[r1 % gU];
  } else { ar0 = r0; ar1 = r1; }
  const f16* arow0 = A + (size_t)ar0 * KT;
  const f16* arow1 = A + (size_t)ar1 * KT;
  const f16* brow[4];
#pragma unroll
  for (int j = 0; j < 4; ++j) brow[j] = Bt + (size_t)(ns * 64 + j * 16 + lw) * KT;

  const v8f vzero = {};
  v8f acc[2][4];
#pragma unroll
  for (int i = 0; i < 2; ++i)
#pragma unroll
    for (int j = 0; j < 4; ++j) acc[i][j] = vzero;

#pragma unroll
  for (int k0 = 0; k0 < KT; k0 += 32) {
    const v16h a0 = load_frag_row(arow0, k0, hi);
    const v16h a1 = load_frag_row(arow1, k0, hi);
    v16h bf[4];
#pragma unroll
    for (int j = 0; j < 4; ++j) bf[j] = load_frag_row(brow[j], k0, hi);
#pragma unroll
    for (int j = 0; j < 4; ++j) {
      acc[0][j] = __builtin_amdgcn_wmma_f32_16x16x32_f16(false, a0, false, bf[j], (short)0, acc[0][j], false, false);
      acc[1][j] = __builtin_amdgcn_wmma_f32_16x16x32_f16(false, a1, false, bf[j], (short)0, acc[1][j], false, false);
    }
  }
  const int mb = ms * 32, nb = ns * 64;
#pragma unroll
  for (int i = 0; i < 2; ++i)
#pragma unroll
    for (int j = 0; j < 4; ++j)
      store_tile<GELU>(acc[i][j], mb + i * 16, nb + j * 16, N, bias, Cf, Ch, lane);
}

// ---------------------------------------------------------------------------
// Weight conversion: f32 RxC row-major -> f16 CxR row-major (transpose)
// ---------------------------------------------------------------------------
__global__ void cvtT_kernel(const float* __restrict__ src, f16* __restrict__ dst, int R, int C) {
  const int i = blockIdx.x * blockDim.x + threadIdx.x;
  if (i >= R * C) return;
  const int r = i / C, c = i % C;
  dst[(size_t)c * R + r] = (f16)src[i];
}

// post[l*T + t] = u such that idx[l][u]==t, else -1
__global__ void build_post_kernel(const int* __restrict__ idx, int* __restrict__ post, int U) {
  const int i = blockIdx.x * blockDim.x + threadIdx.x;
  if (i >= Lc * Tc) return;
  const int l = i / Tc, t = i % Tc;
  int p = -1;
  for (int u = 0; u < U; ++u) if (idx[l * U + u] == t) p = u;
  post[i] = p;
}

// h = x @ proj_w + proj_b + pos_encoding;   writes f32 + f16 copies
__global__ void proj_kernel(const float* __restrict__ x, const float* __restrict__ pw,
                            const float* __restrict__ pb,
                            float* __restrict__ hf, f16* __restrict__ hh) {
  const int bt = blockIdx.x;
  const int d  = threadIdx.x;
  const int t  = bt % Tc;
  const float* xr = x + (size_t)bt * INc;
  float s = pb[d];
#pragma unroll
  for (int k = 0; k < INc; ++k) s += xr[k] * pw[k * Dc + d];
  const int   i2  = d & ~1;
  const float div = __expf((float)i2 * (-9.210340371976184f / (float)Dc));
  const float ang = (float)t * div;
  s += (d & 1) ? __cosf(ang) : __sinf(ang);
  const size_t o = (size_t)bt * Dc + d;
  hf[o] = s; hh[o] = (f16)s;
}

// Attention per (b,h): two-pass softmax over T per selected query u.
__global__ void attention_kernel(const float* __restrict__ qs, const f16* __restrict__ kh,
                                 const f16* __restrict__ vh, f16* __restrict__ ctxc, int U) {
  const int lane = threadIdx.x;
  const int wv   = threadIdx.y;       // 0..7
  const int bh   = blockIdx.x;
  const int b    = bh >> 3, hh = bh & 7;
  const f16* kb = kh + (size_t)b * Tc * Dc + hh * Dh;
  const f16* vb = vh + (size_t)b * Tc * Dc + hh * Dh;
  for (int u = wv; u < U; u += 8) {
    const float* qrow = qs + ((size_t)b * U + u) * Dc + hh * Dh;
    float q[Dh];
#pragma unroll
    for (int j = 0; j < Dh; ++j) q[j] = qrow[j];
    // pass 1: max
    float mx = -1e30f;
    for (int t = lane; t < Tc; t += 32) {
      const v16h kr = *reinterpret_cast<const v16h*>(kb + (size_t)t * Dc);
      float s = 0.f;
#pragma unroll
      for (int j = 0; j < Dh; ++j) s += q[j] * (float)kr[j];
      mx = fmaxf(mx, s * 0.25f);
    }
#pragma unroll
    for (int o = 16; o; o >>= 1) mx = fmaxf(mx, __shfl_xor(mx, o, 32));
    // pass 2: sum + ctx
    float sum = 0.f, c[Dh];
#pragma unroll
    for (int j = 0; j < Dh; ++j) c[j] = 0.f;
    for (int t = lane; t < Tc; t += 32) {
      const v16h kr = *reinterpret_cast<const v16h*>(kb + (size_t)t * Dc);
      const v16h vr = *reinterpret_cast<const v16h*>(vb + (size_t)t * Dc);
      float s = 0.f;
#pragma unroll
      for (int j = 0; j < Dh; ++j) s += q[j] * (float)kr[j];
      const float p = __expf(s * 0.25f - mx);
      sum += p;
#pragma unroll
      for (int j = 0; j < Dh; ++j) c[j] += p * (float)vr[j];
    }
#pragma unroll
    for (int o = 16; o; o >>= 1) {
      sum += __shfl_xor(sum, o, 32);
#pragma unroll
      for (int j = 0; j < Dh; ++j) c[j] += __shfl_xor(c[j], o, 32);
    }
    if (lane == 0) {
      const float inv = 1.0f / sum;
      f16* orow = ctxc + ((size_t)b * U + u) * Dc + hh * Dh;
#pragma unroll
      for (int j = 0; j < Dh; ++j) orow[j] = (f16)(c[j] * inv);
    }
  }
}

// basecat[b][d] = mean_u ctxcat[b][u][d]
__global__ void base_mean_kernel(const f16* __restrict__ ctxc, f16* __restrict__ basec, int U) {
  const int b = blockIdx.x, d = threadIdx.x;
  float s = 0.f;
  for (int u = 0; u < U; ++u) s += (float)ctxc[((size_t)b * U + u) * Dc + d];
  basec[(size_t)b * Dc + d] = (f16)(s / (float)U);
}

// Fused residual + LayerNorm.  addfull!=null -> dense add (ff path);
// else sparse attention add: ctxp at idx rows, rowbase elsewhere.
__global__ void add_ln_kernel(const float* __restrict__ hin, const float* __restrict__ addfull,
                              const float* __restrict__ ctxp, const float* __restrict__ rowb,
                              const int* __restrict__ post, int U,
                              const float* __restrict__ g, const float* __restrict__ bb,
                              float* __restrict__ hout, f16* __restrict__ hh) {
  const int bt = blockIdx.x;
  const int d  = threadIdx.x;
  const int b  = bt / Tc, t = bt % Tc;
  float v = hin[(size_t)bt * Dc + d];
  if (addfull) {
    v += addfull[(size_t)bt * Dc + d];
  } else {
    const int p = post[t];
    v += (p >= 0) ? ctxp[((size_t)b * U + p) * Dc + d] : rowb[(size_t)b * Dc + d];
  }
  __shared__ float sb[Dc];
  sb[d] = v; __syncthreads();
  for (int s = Dc / 2; s > 0; s >>= 1) { if (d < s) sb[d] += sb[d + s]; __syncthreads(); }
  const float mean = sb[0] * (1.0f / Dc);
  __syncthreads();
  const float dv = v - mean;
  sb[d] = dv * dv; __syncthreads();
  for (int s = Dc / 2; s > 0; s >>= 1) { if (d < s) sb[d] += sb[d + s]; __syncthreads(); }
  const float var = sb[0] * (1.0f / Dc);
  const float y = dv * rsqrtf(var + 1e-5f) * g[d] + bb[d];
  const size_t o = (size_t)bt * Dc + d;
  hout[o] = y; hh[o] = (f16)y;
}

// head: LN(last token) @ head_w + head_wb
__global__ void head_kernel(const float* __restrict__ hf, const float* __restrict__ g,
                            const float* __restrict__ bb, const float* __restrict__ hw,
                            const float* __restrict__ hwb, float* __restrict__ out) {
  const int b = blockIdx.x, d = threadIdx.x;
  const float v = hf[((size_t)b * Tc + (Tc - 1)) * Dc + d];
  __shared__ float sb[Dc];
  sb[d] = v; __syncthreads();
  for (int s = Dc / 2; s > 0; s >>= 1) { if (d < s) sb[d] += sb[d + s]; __syncthreads(); }
  const float mean = sb[0] * (1.0f / Dc);
  __syncthreads();
  const float dv = v - mean;
  sb[d] = dv * dv; __syncthreads();
  for (int s = Dc / 2; s > 0; s >>= 1) { if (d < s) sb[d] += sb[d + s]; __syncthreads(); }
  const float var = sb[0] * (1.0f / Dc);
  const float y = dv * rsqrtf(var + 1e-5f) * g[d] + bb[d];
  __syncthreads();
  sb[d] = y * hw[d]; __syncthreads();
  for (int s = Dc / 2; s > 0; s >>= 1) { if (d < s) sb[d] += sb[d + s]; __syncthreads(); }
  if (d == 0) out[b] = sb[0] + hwb[0];
}

// ---------------------------------------------------------------------------
extern "C" void kernel_launch(void* const* d_in, const int* in_sizes, int n_in,
                              void* d_out, int out_size, void* d_ws, size_t ws_size,
                              hipStream_t stream) {
  (void)n_in; (void)out_size; (void)ws_size;
  const float* x      = (const float*)d_in[0];
  const int*   idx    = (const int*)  d_in[1];
  const float* proj_w = (const float*)d_in[2];
  const float* proj_b = (const float*)d_in[3];
  const float* wq     = (const float*)d_in[4];
  const float* wk     = (const float*)d_in[5];
  const float* wvw    = (const float*)d_in[6];
  const float* wo     = (const float*)d_in[7];
  const float* wo_b   = (const float*)d_in[8];
  const float* ln1_g  = (const float*)d_in[9];
  const float* ln1_b  = (const float*)d_in[10];
  const float* ff1_w  = (const float*)d_in[11];
  const float* ff1_b  = (const float*)d_in[12];
  const float* ff2_w  = (const float*)d_in[13];
  const float* ff2_b  = (const float*)d_in[14];
  const float* ln2_g  = (const float*)d_in[15];
  const float* ln2_b  = (const float*)d_in[16];
  const float* head_g = (const float*)d_in[17];
  const float* head_bb= (const float*)d_in[18];
  const float* head_w = (const float*)d_in[19];
  const float* head_wb= (const float*)d_in[20];
  float* out = (float*)d_out;
  const int U = in_sizes[1] / Lc;

  // workspace carve-out (256B aligned)
  char* wsb = (char*)d_ws;
  size_t off = 0;
  auto alloc = [&](size_t bytes) -> char* {
    char* p = wsb + off;
    off = (off + bytes + 255) & ~(size_t)255;
    return p;
  };
  float* hf    = (float*)alloc((size_t)Bc * Tc * Dc * 4);
  f16*   hh    = (f16*)  alloc((size_t)Bc * Tc * Dc * 2);
  f16*   khb   = (f16*)  alloc((size_t)Bc * Tc * Dc * 2);
  f16*   vhb   = (f16*)  alloc((size_t)Bc * Tc * Dc * 2);
  float* qsf   = (float*)alloc((size_t)Bc * U * Dc * 4);
  f16*   ctxc  = (f16*)  alloc((size_t)Bc * U * Dc * 2);
  f16*   basec = (f16*)  alloc((size_t)Bc * Dc * 2);
  float* ctxp  = (float*)alloc((size_t)Bc * U * Dc * 4);
  float* rowb  = (float*)alloc((size_t)Bc * Dc * 4);
  f16*   ff1h  = (f16*)  alloc((size_t)Bc * Tc * 4 * Dc * 2);
  float* ff2f  = (float*)alloc((size_t)Bc * Tc * Dc * 4);
  int*   post  = (int*)  alloc((size_t)Lc * Tc * 4);
  f16*   wqT   = (f16*)  alloc((size_t)Lc * Dc * Dc * 2);
  f16*   wkT   = (f16*)  alloc((size_t)Lc * Dc * Dc * 2);
  f16*   wvT   = (f16*)  alloc((size_t)Lc * Dc * Dc * 2);
  f16*   woT   = (f16*)  alloc((size_t)Lc * Dc * Dc * 2);
  f16*   ff1T  = (f16*)  alloc((size_t)Lc * Dc * 4 * Dc * 2);
  f16*   ff2T  = (f16*)  alloc((size_t)Lc * 4 * Dc * Dc * 2);

  const dim3 gblk(32, 4);
  // K=128 variants
  auto gemm128 = [&](const f16* A, const f16* Bt, const float* bias, float* Cf, f16* Ch,
                     int M, int N, bool gelu, const int* gidx) {
    const int blocks = (M / 32) * (N / 64) / 4;
    if (gelu)
      wmma_gemm_kernel<128, true,  false><<<blocks, gblk, 0, stream>>>(A, Bt, bias, Cf, Ch, M, N, nullptr, 0, 0);
    else if (gidx)
      wmma_gemm_kernel<128, false, true ><<<blocks, gblk, 0, stream>>>(A, Bt, bias, Cf, Ch, M, N, gidx, U, Tc);
    else
      wmma_gemm_kernel<128, false, false><<<blocks, gblk, 0, stream>>>(A, Bt, bias, Cf, Ch, M, N, nullptr, 0, 0);
  };
  auto gemm512 = [&](const f16* A, const f16* Bt, const float* bias, float* Cf, f16* Ch,
                     int M, int N) {
    const int blocks = (M / 32) * (N / 64) / 4;
    wmma_gemm_kernel<512, false, false><<<blocks, gblk, 0, stream>>>(A, Bt, bias, Cf, Ch, M, N, nullptr, 0, 0);
  };

  // weight conversion (f32 KxN -> f16 NxK), per layer
  const int cvB = 256;
  for (int l = 0; l < Lc; ++l) {
    cvtT_kernel<<<(Dc * Dc + cvB - 1) / cvB, cvB, 0, stream>>>(wq  + (size_t)l * Dc * Dc, wqT + (size_t)l * Dc * Dc, Dc, Dc);
    cvtT_kernel<<<(Dc * Dc + cvB - 1) / cvB, cvB, 0, stream>>>(wk  + (size_t)l * Dc * Dc, wkT + (size_t)l * Dc * Dc, Dc, Dc);
    cvtT_kernel<<<(Dc * Dc + cvB - 1) / cvB, cvB, 0, stream>>>(wvw + (size_t)l * Dc * Dc, wvT + (size_t)l * Dc * Dc, Dc, Dc);
    cvtT_kernel<<<(Dc * Dc + cvB - 1) / cvB, cvB, 0, stream>>>(wo  + (size_t)l * Dc * Dc, woT + (size_t)l * Dc * Dc, Dc, Dc);
    cvtT_kernel<<<(Dc * 4 * Dc + cvB - 1) / cvB, cvB, 0, stream>>>(ff1_w + (size_t)l * Dc * 4 * Dc, ff1T + (size_t)l * Dc * 4 * Dc, Dc, 4 * Dc);
    cvtT_kernel<<<(4 * Dc * Dc + cvB - 1) / cvB, cvB, 0, stream>>>(ff2_w + (size_t)l * 4 * Dc * Dc, ff2T + (size_t)l * 4 * Dc * Dc, 4 * Dc, Dc);
  }
  build_post_kernel<<<(Lc * Tc + cvB - 1) / cvB, cvB, 0, stream>>>(idx, post, U);

  // input projection + positional encoding
  proj_kernel<<<Bc * Tc, Dc, 0, stream>>>(x, proj_w, proj_b, hf, hh);

  for (int l = 0; l < Lc; ++l) {
    const size_t wOff = (size_t)l * Dc * Dc;
    // q only at gathered rows; full k, v
    gemm128(hh, wqT + wOff, nullptr, qsf, nullptr, Bc * U,  Dc, false, idx + l * U);
    gemm128(hh, wkT + wOff, nullptr, nullptr, khb, Bc * Tc, Dc, false, nullptr);
    gemm128(hh, wvT + wOff, nullptr, nullptr, vhb, Bc * Tc, Dc, false, nullptr);
    // sparse attention
    attention_kernel<<<Bc * Hc, dim3(32, 8), 0, stream>>>(qsf, khb, vhb, ctxc, U);
    base_mean_kernel<<<Bc, Dc, 0, stream>>>(ctxc, basec, U);
    // out @ wo collapses to: ctx rows + one broadcast row per batch
    gemm128(ctxc,  woT + wOff, wo_b + (size_t)l * Dc, ctxp, nullptr, Bc * U, Dc, false, nullptr);
    gemm128(basec, woT + wOff, wo_b + (size_t)l * Dc, rowb, nullptr, Bc,     Dc, false, nullptr);
    add_ln_kernel<<<Bc * Tc, Dc, 0, stream>>>(hf, nullptr, ctxp, rowb, post + l * Tc, U,
                                              ln1_g + (size_t)l * Dc, ln1_b + (size_t)l * Dc, hf, hh);
    // feed-forward
    gemm128(hh, ff1T + (size_t)l * Dc * 4 * Dc, ff1_b + (size_t)l * 4 * Dc, nullptr, ff1h, Bc * Tc, 4 * Dc, true, nullptr);
    gemm512(ff1h, ff2T + (size_t)l * 4 * Dc * Dc, ff2_b + (size_t)l * Dc, ff2f, nullptr, Bc * Tc, Dc);
    add_ln_kernel<<<Bc * Tc, Dc, 0, stream>>>(hf, ff2f, nullptr, nullptr, nullptr, U,
                                              ln2_g + (size_t)l * Dc, ln2_b + (size_t)l * Dc, hf, hh);
  }
  head_kernel<<<Bc, Dc, 0, stream>>>(hf, head_g, head_bb, head_w, head_wb, out);
}
